// Generator_80367428043228
// MI455X (gfx1250) — matrix operand order repaired and checked by reading, not compiled, not run on hardware
//
#include <hip/hip_runtime.h>
#include <hip/hip_bf16.h>

typedef _Float16 v16h __attribute__((ext_vector_type(16)));
typedef _Float16 v8h  __attribute__((ext_vector_type(8)));
typedef float    v8f  __attribute__((ext_vector_type(8)));

#define NW      8      // waves per block
#define TSTEPS  15
#define HID     64
#define GATES   256    // 4*HID
#define LAT     16
#define HPAD    72     // 64 + 8 halves of pad -> 144B rows (bank stagger, 16B aligned)
#define ZPAD    24     // 16 + 8 halves of pad -> 48B rows

static __device__ __forceinline__ v16h cat8(v8h lo, v8h hi) {
  return __builtin_shufflevector(lo, hi, 0,1,2,3,4,5,6,7,8,9,10,11,12,13,14,15);
}
static __device__ __forceinline__ v8h zero8h() {
  v8h r;
#pragma unroll
  for (int i = 0; i < 8; ++i) r[i] = (_Float16)0.f;
  return r;
}
static __device__ __forceinline__ v8f splat8(float v) {
  v8f r;
#pragma unroll
  for (int i = 0; i < 8; ++i) r[i] = v;
  return r;
}
static __device__ __forceinline__ v8f wmma_f16(v16h a, v16h b, v8f c) {
  // D = A(16x32 f16) * B(32x16 f16) + C(16x16 f32)
  return __builtin_amdgcn_wmma_f32_16x16x32_f16(false, a, false, b, (short)0, c,
                                                false, false);
}
// A fragment (16x32 f16): hb=0 lanes hold K {0..7,16..23}, hb=1 hold {8..15,24..31} (+32*kb)
static __device__ __forceinline__ v16h ld_afrag(const _Float16* buf, int m, int hb, int kb) {
  const _Float16* p = buf + m * HPAD + 32 * kb + 8 * hb;
  return cat8(*(const v8h*)p, *(const v8h*)(p + 16));
}
// B fragment (32x16 f16): lane = column n; hb=0 -> K {0..7,16..23}, hb=1 -> K {8..15,24..31} (+32*kb)
static __device__ __forceinline__ v16h ld_bfrag(const _Float16* w, int n, int hb, int kb) {
  const _Float16* p = w + n * HPAD + 32 * kb + 8 * hb;
  return cat8(*(const v8h*)p, *(const v8h*)(p + 16));
}

// Activations: prefer the gfx1250 V_TANH_F32 transcendental (co-executes with WMMA).
#if __has_builtin(__builtin_amdgcn_tanhf)
static __device__ __forceinline__ float ftanh(float x) { return __builtin_amdgcn_tanhf(x); }
static __device__ __forceinline__ float fsigm(float x) {
  return fmaf(0.5f, __builtin_amdgcn_tanhf(0.5f * x), 0.5f);
}
#else
static __device__ __forceinline__ float ftanh(float x) {
  return 2.0f * __builtin_amdgcn_rcpf(1.0f + __expf(-2.0f * x)) - 1.0f;
}
static __device__ __forceinline__ float fsigm(float x) {
  return __builtin_amdgcn_rcpf(1.0f + __expf(-x));
}
#endif

__global__ __launch_bounds__(NW * 32)
void lstm_gen_kernel(const float* __restrict__ z,
                     const float* __restrict__ Wih0, const float* __restrict__ Whh0,
                     const float* __restrict__ bih0, const float* __restrict__ bhh0,
                     const float* __restrict__ Wih1, const float* __restrict__ Whh1,
                     const float* __restrict__ bih1, const float* __restrict__ bhh1,
                     const float* __restrict__ Wfc,  const float* __restrict__ bfc,
                     float* __restrict__ out)
{
  __shared__ __align__(16) _Float16 w0 [GATES * ZPAD];   // Wih0 rows (K=16)
  __shared__ __align__(16) _Float16 wh0[GATES * HPAD];   // Whh0 rows (K=64)
  __shared__ __align__(16) _Float16 wi1[GATES * HPAD];   // Wih1 rows
  __shared__ __align__(16) _Float16 wh1[GATES * HPAD];   // Whh1 rows
  __shared__ __align__(16) float    b0s[GATES];
  __shared__ __align__(16) float    b1s[GATES];
  __shared__ __align__(16) float    wfcs[HID];
  __shared__ __align__(16) _Float16 zpad16[16];          // zero region for K-padding
  __shared__ __align__(16) _Float16 zbuf [NW][16 * ZPAD];
  __shared__ __align__(16) _Float16 h0buf[NW][16 * HPAD];
  __shared__ __align__(16) _Float16 h1buf[NW][16 * HPAD];
  // xp0 = b0 + z*Wih0^T, time-invariant; f32, D-layout lane-major:
  // tile nt, lane L, component r at xpbuf[wid][(nt*32 + L)*8 + r]  (32B/lane contiguous)
  __shared__ __align__(32) float    xpbuf[NW][16 * 32 * 8];

  const int tid = threadIdx.x;

  // ---- stage weights (f32 -> f16) into LDS, cooperative across the block ----
  for (int i = tid; i < GATES * LAT; i += NW * 32)
    w0[(i >> 4) * ZPAD + (i & 15)] = (_Float16)Wih0[i];
  for (int i = tid; i < GATES * HID; i += NW * 32) {
    const int r = i >> 6, c = i & 63;
    wh0[r * HPAD + c] = (_Float16)Whh0[i];
    wi1[r * HPAD + c] = (_Float16)Wih1[i];
    wh1[r * HPAD + c] = (_Float16)Whh1[i];
  }
  for (int i = tid; i < GATES; i += NW * 32) {
    b0s[i] = bih0[i] + bhh0[i];
    b1s[i] = bih1[i] + bhh1[i];
  }
  for (int i = tid; i < HID; i += NW * 32) wfcs[i] = Wfc[i];
  if (tid < 16) zpad16[tid] = (_Float16)0.f;

  const int lane = tid & 31;
  const int wid  = tid >> 5;
  const int m    = lane & 15;   // row (A) / column (B) handled by this lane
  const int hb   = lane >> 4;   // lane half
  const long base = (long)(blockIdx.x * NW + wid) * 16;

  _Float16* h0b = h0buf[wid];
  _Float16* h1b = h1buf[wid];
  _Float16* zb  = zbuf[wid];
  float*    xpb = xpbuf[wid];
  for (int i = lane; i < 16 * HPAD; i += 32) { h0b[i] = (_Float16)0.f; h1b[i] = (_Float16)0.f; }
  for (int i = lane; i < 16 * LAT; i += 32) {
    const int r = i >> 4, c = i & 15;
    zb[r * ZPAD + c] = (_Float16)z[(base + r) * LAT + c];
  }
  __syncthreads();

  // ---- once per wave: xp0 = b0 + z*Wih0^T (K=16 zero-padded to 32) -> LDS (f32) ----
  {
    const v16h zfrag = cat8(*(const v8h*)(zb + m * ZPAD + 8 * hb), zero8h());
#pragma unroll
    for (int nt = 0; nt < 16; ++nt) {
      const int gi = nt * 16 + m;
      const _Float16* p0 = w0 + gi * ZPAD + 8 * hb;
      const _Float16* p1 = zpad16;
      v8f acc = {};                                   // inline-0 C operand
      acc = wmma_f16(zfrag, cat8(*(const v8h*)p0, *(const v8h*)p1), acc);
      const float bb = b0s[gi];
#pragma unroll
      for (int r = 0; r < 8; ++r) acc[r] += bb;
      *(v8f*)(xpb + (nt * 32 + lane) * 8) = acc;
    }
  }

  // per-lane scalar biases for layer 1 and FC weights (kept as scalars, not tiles)
  float bv1[16];
#pragma unroll
  for (int i = 0; i < 16; ++i) bv1[i] = b1s[i * 16 + m];
  float wfv[4];
#pragma unroll
  for (int n = 0; n < 4; ++n) wfv[n] = wfcs[n * 16 + m];
  const float bfcv = bfc[0];

  v8f c0[4], c1[4];
#pragma unroll
  for (int n = 0; n < 4; ++n) { c0[n] = splat8(0.f); c1[n] = splat8(0.f); }

  for (int t = 0; t < TSTEPS; ++t) {
    // ============ layer 0: gates = xp0 + h0*Whh0^T ============
    const v16h a00 = ld_afrag(h0b, m, hb, 0);   // h0_{t-1}
    const v16h a01 = ld_afrag(h0b, m, hb, 1);
#pragma unroll
    for (int n = 0; n < 4; ++n) {               // column chunk (16 hidden units)
      v8f g[4];
#pragma unroll
      for (int q = 0; q < 4; ++q) {             // gates i, f, g, o
        const int nt = q * 4 + n;
        const int gi = nt * 16 + m;
        v8f acc = *(const v8f*)(xpb + (nt * 32 + lane) * 8);   // xp0 tile as C
        acc = wmma_f16(a00, ld_bfrag(wh0, gi, hb, 0), acc);
        acc = wmma_f16(a01, ld_bfrag(wh0, gi, hb, 1), acc);
        g[q] = acc;
      }
#pragma unroll
      for (int r = 0; r < 8; ++r) {
        const float iv = fsigm(g[0][r]);
        const float fv = fsigm(g[1][r]);
        const float gv = ftanh(g[2][r]);
        const float ov = fsigm(g[3][r]);
        const float cc = fv * c0[n][r] + iv * gv;
        c0[n][r] = cc;
        const float hv = ov * ftanh(cc);
        h0b[(r + 8 * hb) * HPAD + n * 16 + m] = (_Float16)hv;  // D-layout -> row-major LDS
      }
    }
    // ============ layer 1: gates = b1 + h0_t*Wih1^T + h1*Whh1^T ============
    const v16h a10 = ld_afrag(h1b, m, hb, 0);   // h1_{t-1} (read before overwrite)
    const v16h a11 = ld_afrag(h1b, m, hb, 1);
    const v16h x10 = ld_afrag(h0b, m, hb, 0);   // h0_t (same-wave DS ops are in-order)
    const v16h x11 = ld_afrag(h0b, m, hb, 1);
    v8f fcacc = splat8(0.f);
#pragma unroll
    for (int n = 0; n < 4; ++n) {
      v8f g[4];
#pragma unroll
      for (int q = 0; q < 4; ++q) {
        const int nt = q * 4 + n;
        const int gi = nt * 16 + m;
        v8f acc = {};                                          // inline-0 C operand
        acc = wmma_f16(x10, ld_bfrag(wi1, gi, hb, 0), acc);
        acc = wmma_f16(x11, ld_bfrag(wi1, gi, hb, 1), acc);
        acc = wmma_f16(a10, ld_bfrag(wh1, gi, hb, 0), acc);
        acc = wmma_f16(a11, ld_bfrag(wh1, gi, hb, 1), acc);
        g[q] = acc;
      }
#pragma unroll
      for (int r = 0; r < 8; ++r) {
        const float iv = fsigm(g[0][r] + bv1[n]);      // bias folded as scalar adds
        const float fv = fsigm(g[1][r] + bv1[4 + n]);
        const float gv = ftanh(g[2][r] + bv1[8 + n]);
        const float ov = fsigm(g[3][r] + bv1[12 + n]);
        const float cc = fv * c1[n][r] + iv * gv;
        c1[n][r] = cc;
        const float hv = ov * ftanh(cc);
        h1b[(r + 8 * hb) * HPAD + n * 16 + m] = (_Float16)hv;
        fcacc[r] += hv * wfv[n];                       // FC head partial over columns
      }
    }
    // FC reduce over the 16 lanes of each half (stays within the lane half)
#pragma unroll
    for (int mask = 1; mask < 16; mask <<= 1) {
#pragma unroll
      for (int r = 0; r < 8; ++r)
        fcacc[r] += __shfl_xor(fcacc[r], mask, 32);
    }
    if (m == 0) {
#pragma unroll
      for (int r = 0; r < 8; ++r)
        out[(base + r + 8 * hb) * TSTEPS + t] = fcacc[r] + bfcv;
    }
  }
}

extern "C" void kernel_launch(void* const* d_in, const int* in_sizes, int n_in,
                              void* d_out, int out_size, void* d_ws, size_t ws_size,
                              hipStream_t stream) {
  const float* z    = (const float*)d_in[0];
  const float* Wih0 = (const float*)d_in[1];
  const float* Whh0 = (const float*)d_in[2];
  const float* bih0 = (const float*)d_in[3];
  const float* bhh0 = (const float*)d_in[4];
  const float* Wih1 = (const float*)d_in[5];
  const float* Whh1 = (const float*)d_in[6];
  const float* bih1 = (const float*)d_in[7];
  const float* bhh1 = (const float*)d_in[8];
  const float* Wfc  = (const float*)d_in[9];
  const float* bfc  = (const float*)d_in[10];
  float* out = (float*)d_out;

  const int B = in_sizes[0] / LAT;            // 65536
  const int rows_per_block = NW * 16;         // 128
  const int blocks = (B + rows_per_block - 1) / rows_per_block;   // 512
  lstm_gen_kernel<<<dim3(blocks), dim3(NW * 32), 0, stream>>>(
      z, Wih0, Whh0, bih0, bhh0, Wih1, Whh1, bih1, bhh1, Wfc, bfc, out);
}
